// TLocVLBERT_83245056131514
// MI455X (gfx1250) — compile-verified, matrix operand order checked by reading, not run on hardware
//
#include <hip/hip_runtime.h>
#include <hip/hip_bf16.h>

// ---------------------------------------------------------------------------
// TLocVLBERT head on MI455X (gfx1250, wave32, WMMA bf16 -> f32 accum)
// v3: software-pipelined GEMM inner loop (ping-pong fragments) so WMMAs
//     overlap outstanding global_load_b128s; A-operand reuse hints passed as
//     template constants (builtin requires immediate operands).
// ---------------------------------------------------------------------------

typedef __attribute__((ext_vector_type(16))) __bf16 v16bf;
typedef __attribute__((ext_vector_type(8)))  float  v8f;

struct U4 { unsigned int x, y, z, w; };
union Frag { v16bf v; U4 q[2]; };

template <bool RA>
__device__ __forceinline__ v8f wmma_bf16(const v16bf& a, const v16bf& b, v8f c) {
    return __builtin_amdgcn_wmma_f32_16x16x32_bf16(
        false, a, false, b, (short)0, c, /*reuse_a=*/RA, /*reuse_b=*/false);
}

#define B_   32
#define L_   48
#define D_   768
#define T_   32
#define V_   30522
#define H_   768
#define VPAD 30528              // pad V up to multiple of 64 for full WMMA tiles
#define MTXT (B_ * L_)          // 1536 rows of text tokens
#define MOBJ (B_ * (T_ + 1))    // 1056 rows of object tokens (33 per batch)

// ------------------------------- converters --------------------------------

__global__ void f32_to_bf16_kernel(const float* __restrict__ in,
                                   __hip_bfloat16* __restrict__ out, int n) {
    int i = blockIdx.x * blockDim.x + threadIdx.x;
    if (i < n) out[i] = __float2bfloat16(in[i]);
}

// in: K x N row-major f32.  out: Npad x K row-major bf16 (i.e. B column-major),
// zero-filled for n in [N, Npad).  LDS-tiled for coalescing both sides.
__global__ __launch_bounds__(256)
void transpose_to_bf16_kernel(const float* __restrict__ in,
                              __hip_bfloat16* __restrict__ out,
                              int K, int N, int Npad) {
    __shared__ float tile[16][17];
    const int tx = threadIdx.x & 15;
    const int ty = threadIdx.x >> 4;
    const int tn = blockIdx.x * 16;
    const int tk = blockIdx.y * 16;

    int k = tk + ty, n = tn + tx;
    float v = 0.0f;
    if (k < K && n < N) v = in[(size_t)k * N + n];
    tile[ty][tx] = v;
    __syncthreads();

    int on = tn + ty, ok = tk + tx;
    if (on < Npad && ok < K)
        out[(size_t)on * K + ok] = __float2bfloat16(tile[tx][ty]);
}

// ------------------------------ WMMA GEMM ----------------------------------
// C(M,Nreal) = [relu]( A(MxK,bf16 row-major) * Bt(Npad x K bf16, col-major B)
//              + bias )
// One wave -> 16(M) x 64(N) tile via 4 accumulators; 4 waves/block stack in M.
// grid.x = Npad/64, grid.y = ceil(M/16 / 4).  K must be a multiple of 64.
__global__ __launch_bounds__(128)
void gemm_bf16_wmma_kernel(const __hip_bfloat16* __restrict__ A,
                           const __hip_bfloat16* __restrict__ Bt,
                           const float* __restrict__ bias,
                           float* __restrict__ Cf,
                           __hip_bfloat16* __restrict__ Cbf,
                           int M, int Nreal, int K, int ldc, int relu) {
    const int lane  = threadIdx.x & 31;
    const int wave  = threadIdx.x >> 5;
    const int hi    = lane >> 4;     // 0: lanes 0-15, 1: lanes 16-31
    const int llo   = lane & 15;
    const int koff  = hi * 8;        // K sub-chunk select per ISA layout

    const int mtile = blockIdx.y * 4 + wave;
    const int nbase = blockIdx.x * 64;
    const bool mvalid = (mtile * 16) < M;

    int arow = mtile * 16 + llo;
    if (arow >= M) arow = M - 1;     // clamp: keep loads in-bounds, EXEC uniform

    const __hip_bfloat16* Ap = A + (size_t)arow * K;
    const __hip_bfloat16* Bp[4];
#pragma unroll
    for (int j = 0; j < 4; ++j)
        Bp[j] = Bt + (size_t)(nbase + j * 16 + llo) * K;

    v8f acc[4];
#pragma unroll
    for (int j = 0; j < 4; ++j) acc[j] = (v8f){};

    // ---- 2-stage software pipeline: prefetch k+32 before computing k ----
    Frag a0, a1, b0[4], b1[4];

    a0.q[0] = *reinterpret_cast<const U4*>(Ap + koff);
    a0.q[1] = *reinterpret_cast<const U4*>(Ap + 16 + koff);
#pragma unroll
    for (int j = 0; j < 4; ++j) {
        b0[j].q[0] = *reinterpret_cast<const U4*>(Bp[j] + koff);
        b0[j].q[1] = *reinterpret_cast<const U4*>(Bp[j] + 16 + koff);
    }

    for (int k0 = 0; k0 < K; k0 += 64) {
        const int k1 = k0 + 32;
        a1.q[0] = *reinterpret_cast<const U4*>(Ap + k1 + koff);
        a1.q[1] = *reinterpret_cast<const U4*>(Ap + k1 + 16 + koff);
#pragma unroll
        for (int j = 0; j < 4; ++j) {
            b1[j].q[0] = *reinterpret_cast<const U4*>(Bp[j] + k1 + koff);
            b1[j].q[1] = *reinterpret_cast<const U4*>(Bp[j] + k1 + 16 + koff);
        }
        acc[0] = wmma_bf16<true >(a0.v, b0[0].v, acc[0]);
        acc[1] = wmma_bf16<true >(a0.v, b0[1].v, acc[1]);
        acc[2] = wmma_bf16<true >(a0.v, b0[2].v, acc[2]);
        acc[3] = wmma_bf16<false>(a0.v, b0[3].v, acc[3]);

        const int k2 = (k0 + 64 < K) ? k0 + 64 : 0;   // dummy-safe last prefetch
        a0.q[0] = *reinterpret_cast<const U4*>(Ap + k2 + koff);
        a0.q[1] = *reinterpret_cast<const U4*>(Ap + k2 + 16 + koff);
#pragma unroll
        for (int j = 0; j < 4; ++j) {
            b0[j].q[0] = *reinterpret_cast<const U4*>(Bp[j] + k2 + koff);
            b0[j].q[1] = *reinterpret_cast<const U4*>(Bp[j] + k2 + 16 + koff);
        }
        acc[0] = wmma_bf16<true >(a1.v, b1[0].v, acc[0]);
        acc[1] = wmma_bf16<true >(a1.v, b1[1].v, acc[1]);
        acc[2] = wmma_bf16<true >(a1.v, b1[2].v, acc[2]);
        acc[3] = wmma_bf16<false>(a1.v, b1[3].v, acc[3]);
    }

    if (!mvalid) return;             // wave-uniform, after all WMMAs
#pragma unroll
    for (int j = 0; j < 4; ++j) {
        const int col = nbase + j * 16 + llo;
        if (col >= Nreal) continue;
        const float bv = bias ? bias[col] : 0.0f;
#pragma unroll
        for (int r = 0; r < 8; ++r) {
            const int row = mtile * 16 + r + 8 * hi;
            if (row >= M) continue;
            float v = acc[j][r] + bv;
            if (relu) v = fmaxf(v, 0.0f);
            if (Cf)  Cf [(size_t)row * ldc + col] = v;
            else     Cbf[(size_t)row * ldc + col] = __float2bfloat16(v);
        }
    }
}

// --------------------------- IoU combine stage -----------------------------
// h[b,s,e,:] = relu(hs[b,s,:] + hcp[b,(s+e)/2,:] + he[b,e,:] + b_i1)
// logits_iou[b,k,s,e] = h . w_i2[:,k] + b_i2[k]
// One wave per (b,s,e); lane-strided 768-feature reduction, shuffle tree.
__global__ __launch_bounds__(128)
void iou_combine_kernel(const float* __restrict__ hs,
                        const float* __restrict__ hcp,
                        const float* __restrict__ he,
                        const float* __restrict__ b1,
                        const float* __restrict__ w2,
                        const float* __restrict__ b2,
                        float* __restrict__ out) {
    const int gw   = blockIdx.x * 4 + (threadIdx.x >> 5);
    const int lane = threadIdx.x & 31;
    const int b = gw / (T_ * (T_ + 1));
    const int r = gw - b * (T_ * (T_ + 1));
    const int s = r / (T_ + 1);
    const int e = r - s * (T_ + 1);
    const int c = (s + e) >> 1;

    const float* ps = hs  + (size_t)(b * (T_ + 1) + s) * H_;
    const float* pc = hcp + (size_t)(b * (T_ + 1) + c) * H_;
    const float* pe = he  + (size_t)(b * (T_ + 1) + e) * H_;

    float a0 = 0.f, a1 = 0.f, a2 = 0.f;
    for (int f = lane; f < H_; f += 32) {
        float v = ps[f] + pc[f] + pe[f] + b1[f];
        v = fmaxf(v, 0.0f);
        a0 += v * w2[f * 3 + 0];
        a1 += v * w2[f * 3 + 1];
        a2 += v * w2[f * 3 + 2];
    }
#pragma unroll
    for (int off = 16; off > 0; off >>= 1) {
        a0 += __shfl_down(a0, off, 32);
        a1 += __shfl_down(a1, off, 32);
        a2 += __shfl_down(a2, off, 32);
    }
    if (lane == 0) {
        const size_t se = (size_t)s * (T_ + 1) + e;
        out[((size_t)(b * 3 + 0) * T_ * (T_ + 1)) + se] = a0 + b2[0];
        out[((size_t)(b * 3 + 1) * T_ * (T_ + 1)) + se] = a1 + b2[1];
        out[((size_t)(b * 3 + 2) * T_ * (T_ + 1)) + se] = a2 + b2[2];
    }
}

// ------------------------------- mask map ----------------------------------

__global__ void iou_mask_kernel(float* __restrict__ out) {
    int idx = blockIdx.x * blockDim.x + threadIdx.x;
    if (idx >= 32 * 33) return;
    int i = idx / 33, j = idx % 33;
    float v = 0.0f;
    int hi = (i + 17 < 33) ? (i + 17) : 33;
    if (j >= i + 1 && j < hi) v = 1.0f;
    if (i < 16 && (i & 1) == 0 && j >= 18 + i && (((j - (18 + i)) & 1) == 0))
        v = 1.0f;
    out[idx] = v;
}

// ------------------------------- launcher ----------------------------------

extern "C" void kernel_launch(void* const* d_in, const int* in_sizes, int n_in,
                              void* d_out, int out_size, void* d_ws, size_t ws_size,
                              hipStream_t stream) {
    (void)in_sizes; (void)n_in; (void)out_size; (void)ws_size;

    const float* x_text = (const float*)d_in[0];   // (32,48,768)
    const float* x_obj  = (const float*)d_in[1];   // (32,33,768)
    const float* w_t1   = (const float*)d_in[2];   // (768,768)
    const float* b_t1   = (const float*)d_in[3];   // (768)
    const float* w_t2   = (const float*)d_in[4];   // (768,30522)
    const float* b_t2   = (const float*)d_in[5];   // (30522)
    const float* w_i1   = (const float*)d_in[6];   // (2304,768)
    const float* b_i1   = (const float*)d_in[7];   // (768)
    const float* w_i2   = (const float*)d_in[8];   // (768,3)
    const float* b_i2   = (const float*)d_in[9];   // (3)

    float* out_text = (float*)d_out;                                   // 1536*30522
    float* out_iou  = out_text + (size_t)MTXT * V_;                    // 32*3*32*33
    float* out_mask = out_iou + (size_t)B_ * 3 * T_ * (T_ + 1);        // 32*33

    // ---- workspace layout (bytes, all regions 256B aligned) ----
    char* ws = (char*)d_ws;
    size_t off = 0;
    auto take = [&](size_t bytes) { char* p = ws + off; off += (bytes + 255) & ~(size_t)255; return p; };
    __hip_bfloat16* xt_bf  = (__hip_bfloat16*)take((size_t)MTXT * D_ * 2);  // text acts, bf16
    __hip_bfloat16* ht_bf  = (__hip_bfloat16*)take((size_t)MTXT * H_ * 2);  // relu hidden, bf16
    __hip_bfloat16* hc_bf  = (__hip_bfloat16*)take((size_t)MOBJ * D_ * 2);  // object acts, bf16
    __hip_bfloat16* wt1T   = (__hip_bfloat16*)take((size_t)H_ * D_ * 2);    // w_t1^T
    __hip_bfloat16* wi1T   = (__hip_bfloat16*)take((size_t)3 * H_ * D_ * 2);// 3 blocks of w_i1^T
    float*          hs     = (float*)take((size_t)MOBJ * H_ * 4);           // hc @ W_s
    float*          hcp    = (float*)take((size_t)MOBJ * H_ * 4);           // hc @ W_c
    float*          he     = (float*)take((size_t)MOBJ * H_ * 4);           // hc @ W_e
    __hip_bfloat16* wt2T   = (__hip_bfloat16*)take((size_t)VPAD * H_ * 2);  // w_t2^T, zero-padded

    // ---- stage 0: precision conversion + weight transposes ----
    {
        int n = MTXT * D_;
        f32_to_bf16_kernel<<<(n + 255) / 256, 256, 0, stream>>>(x_text, xt_bf, n);
        n = MOBJ * D_;
        f32_to_bf16_kernel<<<(n + 255) / 256, 256, 0, stream>>>(x_obj, hc_bf, n);
    }
    {
        dim3 blk(256);
        transpose_to_bf16_kernel<<<dim3(H_ / 16, D_ / 16), blk, 0, stream>>>(
            w_t1, wt1T, D_, H_, H_);
        transpose_to_bf16_kernel<<<dim3(VPAD / 16, H_ / 16), blk, 0, stream>>>(
            w_t2, wt2T, H_, V_, VPAD);
        for (int j = 0; j < 3; ++j)
            transpose_to_bf16_kernel<<<dim3(H_ / 16, D_ / 16), blk, 0, stream>>>(
                w_i1 + (size_t)j * D_ * H_, wi1T + (size_t)j * H_ * D_, D_, H_, H_);
    }

    // ---- stage 1: text hidden = relu(X @ w_t1 + b_t1), kept in bf16 ----
    gemm_bf16_wmma_kernel<<<dim3(H_ / 64, (MTXT / 16 + 3) / 4), 128, 0, stream>>>(
        xt_bf, wt1T, b_t1, nullptr, ht_bf, MTXT, H_, D_, H_, /*relu=*/1);

    // ---- stage 2: logits_text = hidden @ w_t2 + b_t2  (the 72-GFLOP GEMM) ----
    gemm_bf16_wmma_kernel<<<dim3(VPAD / 64, (MTXT / 16 + 3) / 4), 128, 0, stream>>>(
        ht_bf, wt2T, b_t2, out_text, nullptr, MTXT, V_, H_, V_, /*relu=*/0);

    // ---- stage 3: three object projections (factorized sce @ w_i1) ----
    {
        dim3 grid(H_ / 64, ((MOBJ + 15) / 16 + 3) / 4);
        gemm_bf16_wmma_kernel<<<grid, 128, 0, stream>>>(
            hc_bf, wi1T + 0 * (size_t)H_ * D_, nullptr, hs,  nullptr, MOBJ, H_, D_, H_, 0);
        gemm_bf16_wmma_kernel<<<grid, 128, 0, stream>>>(
            hc_bf, wi1T + 1 * (size_t)H_ * D_, nullptr, hcp, nullptr, MOBJ, H_, D_, H_, 0);
        gemm_bf16_wmma_kernel<<<grid, 128, 0, stream>>>(
            hc_bf, wi1T + 2 * (size_t)H_ * D_, nullptr, he,  nullptr, MOBJ, H_, D_, H_, 0);
    }

    // ---- stage 4: gather + relu + (768 -> 3) contraction ----
    {
        int waves = B_ * T_ * (T_ + 1);  // 33792, divisible by 4
        iou_combine_kernel<<<waves / 4, 128, 0, stream>>>(
            hs, hcp, he, b_i1, w_i2, b_i2, out_iou);
    }

    // ---- stage 5: static mask map ----
    iou_mask_kernel<<<(32 * 33 + 127) / 128, 128, 0, stream>>>(out_mask);
}
